// PyraformerLRModel_49082886259377
// MI455X (gfx1250) — matrix operand order repaired and checked by reading, not compiled, not run on hardware
//
#include <hip/hip_runtime.h>
#include <hip/hip_bf16.h>
#include <math.h>

// ---------------------------------------------------------------------------
// Types for WMMA fragments (CDNA5 / gfx1250, wave32)
// ---------------------------------------------------------------------------
typedef __attribute__((ext_vector_type(16))) _Float16 v16h;
typedef __attribute__((ext_vector_type(8)))  _Float16 v8h;
typedef __attribute__((ext_vector_type(8)))  float    v8f;

#define SEQ_   512
#define NTOK_  680
#define DM_    256
#define BATCH_ 32
#define MROWS_ (BATCH_ * NTOK_)      // 21760
#define HEADS_ 4
#define DK_    64

// ---------------------------------------------------------------------------
// Helpers
// ---------------------------------------------------------------------------
__device__ __forceinline__ float block_reduce_sum_256(float v, float* sm) {
  // 256 threads = 8 waves (wave32)
  #pragma unroll
  for (int o = 16; o >= 1; o >>= 1) v += __shfl_xor(v, o, 32);
  int w = threadIdx.x >> 5;
  if ((threadIdx.x & 31) == 0) sm[w] = v;
  __syncthreads();
  float t = 0.f;
  #pragma unroll
  for (int i = 0; i < 8; ++i) t += sm[i];
  __syncthreads();
  return t;
}

// ---------------------------------------------------------------------------
// 1) Embedding: circular conv(k=3) + time-linear + fixed positional encoding
//    emb[b,t,d], grid = B*SEQ blocks, 256 threads (d)
// ---------------------------------------------------------------------------
__global__ void embed_kernel(const float* __restrict__ x_enc,   // [B,512,1]
                             const float* __restrict__ x_mark,  // [B,512,4]
                             const float* __restrict__ W_val,   // [256,1,3]
                             const float* __restrict__ W_time,  // [4,256]
                             const float* __restrict__ b_time,  // [256]
                             float* __restrict__ emb)           // [B,512,256]
{
  int row = blockIdx.x;                // b*512 + t
  int t   = row % SEQ_;
  int b   = row / SEQ_;
  int d   = threadIdx.x;

  // circular conv
  int tm = (t + SEQ_ - 1) % SEQ_;
  int tp = (t + 1) % SEQ_;
  float xm1 = x_enc[(size_t)b * SEQ_ + tm];
  float x0  = x_enc[(size_t)b * SEQ_ + t];
  float xp1 = x_enc[(size_t)b * SEQ_ + tp];
  float acc = xm1 * W_val[d * 3 + 0] + x0 * W_val[d * 3 + 1] + xp1 * W_val[d * 3 + 2];

  // time features
  const float* mk = x_mark + ((size_t)b * SEQ_ + t) * 4;
  #pragma unroll
  for (int c = 0; c < 4; ++c) acc += mk[c] * W_time[c * DM_ + d];
  acc += b_time[d];

  // positional encoding (sin/cos), div = exp(-(d_even) * ln(10000)/256)
  int de = d & ~1;
  float div = __expf(-(float)de * (9.210340371976184f / (float)DM_));
  float arg = (float)t * div;
  acc += (d & 1) ? cosf(arg) : sinf(arg);

  emb[(size_t)row * DM_ + d] = acc;
}

// ---------------------------------------------------------------------------
// 2) Weight convert + transpose: W[K][N] f32 -> Wt[N][K] f16
// ---------------------------------------------------------------------------
__global__ void wcvt_t_kernel(const float* __restrict__ W, _Float16* __restrict__ Wt,
                              int K, int N)
{
  int idx = blockIdx.x * blockDim.x + threadIdx.x;
  if (idx >= K * N) return;
  int k = idx % K;
  int n = idx / K;
  Wt[(size_t)n * K + k] = (_Float16)W[(size_t)k * N + n];
}

// ---------------------------------------------------------------------------
// 3) Generic WMMA GEMM: C[M,N] = A[M,K](f32) * Bt[N,K](f16)^T + bias, opt ReLU
//    One wave -> 16x64 tile via 4x v_wmma_f32_16x16x32_f16. 8 waves / block.
// ---------------------------------------------------------------------------
__global__ __launch_bounds__(256)
void gemm_wmma_kernel(const float* __restrict__ A, const _Float16* __restrict__ Bt,
                      const float* __restrict__ bias, float* __restrict__ C,
                      int M, int K, int N, int relu)
{
  int wave = (blockIdx.x * 256 + threadIdx.x) >> 5;
  int lane = threadIdx.x & 31;
  int ntilesN = N >> 6;
  int tiles   = (M >> 4) * ntilesN;
  if (wave >= tiles) return;                 // wave-uniform exit, EXEC stays full
  int mt = wave / ntilesN;
  int nt = wave % ntilesN;

  int mrow  = (lane & 15);                   // A: lane -> M row
  int khalf = lane >> 4;                     // A: lane group -> K offset 0/8
  int kgrp  = (lane >> 4) * 8;               // B: lane group -> K offset 0/8
  const float* arow = A + ((size_t)(mt * 16 + mrow)) * K + khalf * 8;
  int ncol0 = nt * 64;

  v8f acc[4] = {};

  for (int kk = 0; kk < K; kk += 32) {
    // ---- A fragment: 16x32 f16 (ISA 7.12.2: lanes 0-15 K=0..7/16..23, 16-31 +8)
    v16h a;
    {
      const float4* p0 = reinterpret_cast<const float4*>(arow + kk);
      const float4* p1 = reinterpret_cast<const float4*>(arow + kk + 16);
      float4 f0 = p0[0], f1 = p0[1];
      float4 g0 = p1[0], g1 = p1[1];
      a[0]  = (_Float16)f0.x; a[1]  = (_Float16)f0.y; a[2]  = (_Float16)f0.z; a[3]  = (_Float16)f0.w;
      a[4]  = (_Float16)f1.x; a[5]  = (_Float16)f1.y; a[6]  = (_Float16)f1.z; a[7]  = (_Float16)f1.w;
      a[8]  = (_Float16)g0.x; a[9]  = (_Float16)g0.y; a[10] = (_Float16)g0.z; a[11] = (_Float16)g0.w;
      a[12] = (_Float16)g1.x; a[13] = (_Float16)g1.y; a[14] = (_Float16)g1.z; a[15] = (_Float16)g1.w;
    }
    // ---- 4 B fragments (32x16 f16 each), reuse A
    #pragma unroll
    for (int sub = 0; sub < 4; ++sub) {
      const v8h* bp = reinterpret_cast<const v8h*>(
          Bt + ((size_t)(ncol0 + sub * 16 + (lane & 15))) * K + kk + kgrp);
      v8h lo = bp[0], hi = bp[2];
      v16h bf = __builtin_shufflevector(lo, hi, 0,1,2,3,4,5,6,7,8,9,10,11,12,13,14,15);
      acc[sub] = __builtin_amdgcn_wmma_f32_16x16x32_f16(
          false, a, false, bf, (short)0, acc[sub], false, false);
    }
  }

  // ---- Epilogue (C/D layout: VGPR r -> M = r + 8*(lane>=16), N = lane&15)
  int mg = lane >> 4;
  #pragma unroll
  for (int sub = 0; sub < 4; ++sub) {
    int n = ncol0 + sub * 16 + (lane & 15);
    float bv = bias ? bias[n] : 0.f;
    #pragma unroll
    for (int r = 0; r < 8; ++r) {
      float v = acc[sub][r] + bv;
      if (relu) v = fmaxf(v, 0.f);
      C[((size_t)(mt * 16 + mg * 8 + r)) * N + n] = v;
    }
  }
}

// ---------------------------------------------------------------------------
// 4) CSCM strided conv (k=4, stride=4) + bias + BN affine + ELU
// ---------------------------------------------------------------------------
__global__ void cscm_conv_kernel(const float* __restrict__ in, size_t in_bstride,
                                 const float* __restrict__ Wc,   // [64][64][4]
                                 const float* __restrict__ bc,
                                 const float* __restrict__ bg,
                                 const float* __restrict__ bb,
                                 float* __restrict__ out, size_t out_bstride,
                                 int Lout)
{
  int idx = blockIdx.x * blockDim.x + threadIdx.x;
  if (idx >= BATCH_ * Lout * 64) return;
  int c = idx & 63;
  int r = idx >> 6;
  int j = r % Lout;
  int b = r / Lout;
  const float* ip = in + (size_t)b * in_bstride + (size_t)(j * 4) * 64;
  float acc = 0.f;
  #pragma unroll 4
  for (int ic = 0; ic < 64; ++ic) {
    #pragma unroll
    for (int h = 0; h < 4; ++h)
      acc += ip[h * 64 + ic] * Wc[(c * 64 + ic) * 4 + h];
  }
  acc = (acc + bc[c]) * bg[c] + bb[c];
  acc = (acc > 0.f) ? acc : (expm1f(acc));     // ELU(alpha=1)
  out[(size_t)b * out_bstride + (size_t)j * 64 + c] = acc;
}

// ---------------------------------------------------------------------------
// 5) concat [emb ; up-proj pooled] + LayerNorm  -> x [B*680, 256]
// ---------------------------------------------------------------------------
__global__ void concat_ln_kernel(const float* __restrict__ emb,
                                 const float* __restrict__ up,
                                 const float* __restrict__ g,
                                 const float* __restrict__ b,
                                 float* __restrict__ x)
{
  __shared__ float sm[8];
  int row = blockIdx.x;               // b*680 + t
  int t = row % NTOK_, bb = row / NTOK_;
  int d = threadIdx.x;
  float val = (t < SEQ_)
      ? emb[((size_t)bb * SEQ_ + t) * DM_ + d]
      : up[((size_t)bb * 168 + (t - SEQ_)) * DM_ + d];
  float mean = block_reduce_sum_256(val, sm) * (1.f / 256.f);
  float dv = val - mean;
  float var = block_reduce_sum_256(dv * dv, sm) * (1.f / 256.f);
  x[(size_t)row * DM_ + d] = dv * rsqrtf(var + 1e-5f) * g[d] + b[d];
}

// ---------------------------------------------------------------------------
// 6) Residual add + LayerNorm (post-norm)
// ---------------------------------------------------------------------------
__global__ void add_ln_kernel(const float* __restrict__ xin,
                              const float* __restrict__ delta,
                              const float* __restrict__ g,
                              const float* __restrict__ b,
                              float* __restrict__ xout)
{
  __shared__ float sm[8];
  int row = blockIdx.x;
  int d = threadIdx.x;
  float val = xin[(size_t)row * DM_ + d] + delta[(size_t)row * DM_ + d];
  float mean = block_reduce_sum_256(val, sm) * (1.f / 256.f);
  float dv = val - mean;
  float var = block_reduce_sum_256(dv * dv, sm) * (1.f / 256.f);
  xout[(size_t)row * DM_ + d] = dv * rsqrtf(var + 1e-5f) * g[d] + b[d];
}

// ---------------------------------------------------------------------------
// 7) Pyraformer adjacency (mask is ~98.5% sparse: band<=5 + 4 children + parent)
// ---------------------------------------------------------------------------
__global__ void adj_build_kernel(int* __restrict__ adj, int* __restrict__ cnt)
{
  int q = blockIdx.x * blockDim.x + threadIdx.x;
  if (q >= NTOK_) return;
  const int starts[5] = {0, 512, 640, 672, 680};
  const int sizes[4]  = {512, 128, 32, 8};
  int lq = 0;
  while (lq < 3 && q >= starts[lq + 1]) lq++;
  int st = starts[lq], sz = sizes[lq];
  int c = 0;
  int lo = q - 2; if (lo < st) lo = st;
  int hi = q + 2; if (hi > st + sz - 1) hi = st + sz - 1;
  for (int j = lo; j <= hi; ++j) adj[q * 12 + c++] = j;          // intra-scale band
  if (lq >= 1) {                                                 // children
    int cb = starts[lq - 1] + (q - st) * 4;
    #pragma unroll
    for (int h = 0; h < 4; ++h) adj[q * 12 + c++] = cb + h;
  }
  if (lq <= 2) adj[q * 12 + c++] = starts[lq + 1] + (q - st) / 4; // parent
  cnt[q] = c;
  while (c < 12) adj[q * 12 + c++] = -1;
}

// ---------------------------------------------------------------------------
// 8) Sparse masked attention: one wave per (b,h,q); 32 lanes cover 64 dims.
//    exp(-1e9) underflows to 0 in f32, so allowed-only softmax == reference.
// ---------------------------------------------------------------------------
__global__ __launch_bounds__(256)
void attn_kernel(const float* __restrict__ q, const float* __restrict__ k,
                 const float* __restrict__ v, const int* __restrict__ adj,
                 const int* __restrict__ cnt, float* __restrict__ o)
{
  int wid  = (blockIdx.x * 256 + threadIdx.x) >> 5;
  int lane = threadIdx.x & 31;
  if (wid >= BATCH_ * HEADS_ * NTOK_) return;
  int t  = wid % NTOK_;
  int bh = wid / NTOK_;
  int h  = bh & (HEADS_ - 1);
  int b  = bh / HEADS_;
  size_t row = (size_t)b * NTOK_ + t;
  int col0 = h * DK_ + lane * 2;

  float2 qv = *reinterpret_cast<const float2*>(q + row * DM_ + col0);
  int n = cnt[t];                       // wave-uniform
  float s[12];
  float mx = -1e30f;
  #pragma unroll
  for (int i = 0; i < 12; ++i) {
    if (i < n) {
      int j = adj[t * 12 + i];
      float2 kv = *reinterpret_cast<const float2*>(k + ((size_t)b * NTOK_ + j) * DM_ + col0);
      float p = qv.x * kv.x + qv.y * kv.y;
      #pragma unroll
      for (int off = 16; off >= 1; off >>= 1) p += __shfl_xor(p, off, 32);
      p *= 0.125f;                      // 1/sqrt(64)
      s[i] = p;
      mx = fmaxf(mx, p);
    } else {
      s[i] = -1e30f;
    }
  }
  float sum = 0.f;
  #pragma unroll
  for (int i = 0; i < 12; ++i) {
    float e = (i < n) ? __expf(s[i] - mx) : 0.f;
    s[i] = e;
    sum += e;
  }
  float inv = 1.f / sum;
  float ax = 0.f, ay = 0.f;
  #pragma unroll
  for (int i = 0; i < 12; ++i) {
    if (i < n) {
      int j = adj[t * 12 + i];
      float2 vv = *reinterpret_cast<const float2*>(v + ((size_t)b * NTOK_ + j) * DM_ + col0);
      float w = s[i] * inv;
      ax += w * vv.x;
      ay += w * vv.y;
    }
  }
  float2 res = {ax, ay};
  *reinterpret_cast<float2*>(o + row * DM_ + col0) = res;
}

// ---------------------------------------------------------------------------
// 9) Final decoder: gather nodes {511,639,671,679}, concat, @ Wp + bp
// ---------------------------------------------------------------------------
__global__ void final_proj_kernel(const float* __restrict__ x,
                                  const float* __restrict__ Wp,  // [1024,96]
                                  const float* __restrict__ bp,  // [96]
                                  float* __restrict__ out)       // [B,96]
{
  int j = threadIdx.x;                 // 96
  int b = blockIdx.x;                  // 32
  if (j >= 96) return;
  const int IDX[4] = {511, 639, 671, 679};
  float acc = bp[j];
  #pragma unroll
  for (int s = 0; s < 4; ++s) {
    const float* xr = x + ((size_t)b * NTOK_ + IDX[s]) * DM_;
    for (int d = 0; d < DM_; ++d)
      acc += xr[d] * Wp[(size_t)(s * DM_ + d) * 96 + j];
  }
  out[(size_t)b * 96 + j] = acc;
}

// ---------------------------------------------------------------------------
// Host launcher
// ---------------------------------------------------------------------------
extern "C" void kernel_launch(void* const* d_in, const int* in_sizes, int n_in,
                              void* d_out, int out_size, void* d_ws, size_t ws_size,
                              hipStream_t stream)
{
  (void)in_sizes; (void)n_in; (void)out_size;
  const float* x_enc  = (const float*)d_in[0];
  const float* x_mark = (const float*)d_in[1];
  const float* W_val  = (const float*)d_in[2];
  const float* W_time = (const float*)d_in[3];
  const float* b_time = (const float*)d_in[4];
  const float* W_down = (const float*)d_in[5];
  const float* b_down = (const float*)d_in[6];
  const float* W_conv = (const float*)d_in[7];
  const float* b_conv = (const float*)d_in[8];
  const float* bn_g   = (const float*)d_in[9];
  const float* bn_b   = (const float*)d_in[10];
  const float* W_up   = (const float*)d_in[11];
  const float* b_up   = (const float*)d_in[12];
  const float* ln_c_g = (const float*)d_in[13];
  const float* ln_c_b = (const float*)d_in[14];
  const float* Wq     = (const float*)d_in[15];
  const float* bq     = (const float*)d_in[16];
  const float* Wk     = (const float*)d_in[17];
  const float* bk     = (const float*)d_in[18];
  const float* Wv     = (const float*)d_in[19];
  const float* bv     = (const float*)d_in[20];
  const float* Wo     = (const float*)d_in[21];
  const float* bo     = (const float*)d_in[22];
  const float* ln1_g  = (const float*)d_in[23];
  const float* ln1_b  = (const float*)d_in[24];
  const float* W1f    = (const float*)d_in[25];
  const float* b1f    = (const float*)d_in[26];
  const float* W2f    = (const float*)d_in[27];
  const float* b2f    = (const float*)d_in[28];
  const float* ln2_g  = (const float*)d_in[29];
  const float* ln2_b  = (const float*)d_in[30];
  const float* Wp     = (const float*)d_in[31];
  const float* bp     = (const float*)d_in[32];
  float* out = (float*)d_out;

  // ---- workspace layout (with aggressive reuse) ----
  char* base = (char*)d_ws;
  size_t off = 0;
  auto take = [&](size_t bytes) { size_t r = off; off += (bytes + 255) & ~(size_t)255; return r; };
  const size_t ACT = (size_t)MROWS_ * DM_ * 4;          // 22,282,240 B
  size_t x_off   = take(ACT);
  size_t r12_off = take((size_t)MROWS_ * 512 * 4);      // q | k ; reused as f1
  size_t r3_off  = take(ACT);                           // v ; f2 ; pool+up_out
  size_t r4_off  = take(ACT);                           // attn_o ; emb
  size_t r5_off  = take(ACT);                           // proj ; t0 (down-proj)
  size_t adj_off = take(680 * 12 * 4);
  size_t cnt_off = take(680 * 4);
  size_t wdt_off = take((size_t)256 * 64 * 2);
  size_t wut_off = take((size_t)64 * 256 * 2);
  size_t wq_off  = take((size_t)2 * 256 * 256 * 2);
  size_t wk_off  = take((size_t)2 * 256 * 256 * 2);
  size_t wv_off  = take((size_t)2 * 256 * 256 * 2);
  size_t wo_off  = take((size_t)2 * 256 * 256 * 2);
  size_t w1_off  = take((size_t)2 * 256 * 512 * 2);
  size_t w2_off  = take((size_t)2 * 512 * 256 * 2);
  if (ws_size < off) return;  // not enough scratch; bail safely

  float* x      = (float*)(base + x_off);
  float* qbuf   = (float*)(base + r12_off);
  float* kbuf   = (float*)(base + r12_off + ACT);
  float* f1     = (float*)(base + r12_off);             // overlays q+k
  float* vbuf   = (float*)(base + r3_off);
  float* f2     = (float*)(base + r3_off);              // overlays v
  float* pool   = (float*)(base + r3_off);              // phase-A only
  float* up_out = (float*)(base + r3_off + 0x160000);   // 1.44MB past pool
  float* attn_o = (float*)(base + r4_off);
  float* emb    = (float*)(base + r4_off);              // phase-A only
  float* proj   = (float*)(base + r5_off);
  float* t0     = (float*)(base + r5_off);              // phase-A only
  int*   adj    = (int*)(base + adj_off);
  int*   cnt    = (int*)(base + cnt_off);
  _Float16* wdt = (_Float16*)(base + wdt_off);
  _Float16* wut = (_Float16*)(base + wut_off);
  _Float16* wqt = (_Float16*)(base + wq_off);
  _Float16* wkt = (_Float16*)(base + wk_off);
  _Float16* wvt = (_Float16*)(base + wv_off);
  _Float16* wot = (_Float16*)(base + wo_off);
  _Float16* w1t = (_Float16*)(base + w1_off);
  _Float16* w2t = (_Float16*)(base + w2_off);

  auto cvt = [&](const float* W, _Float16* Wt, int K, int N) {
    int tot = K * N;
    wcvt_t_kernel<<<(tot + 255) / 256, 256, 0, stream>>>(W, Wt, K, N);
  };
  auto gemm = [&](const float* A, const _Float16* Bt, const float* bias, float* C,
                  int M, int K, int N, int relu) {
    int tiles = (M / 16) * (N / 64);
    gemm_wmma_kernel<<<(tiles + 7) / 8, 256, 0, stream>>>(A, Bt, bias, C, M, K, N, relu);
  };

  // ---- weight preprocessing (small, once per launch) ----
  cvt(W_down, wdt, 256, 64);
  cvt(W_up,   wut, 64, 256);
  for (int l = 0; l < 2; ++l) {
    cvt(Wq + (size_t)l * 65536, wqt + (size_t)l * 65536, 256, 256);
    cvt(Wk + (size_t)l * 65536, wkt + (size_t)l * 65536, 256, 256);
    cvt(Wv + (size_t)l * 65536, wvt + (size_t)l * 65536, 256, 256);
    cvt(Wo + (size_t)l * 65536, wot + (size_t)l * 65536, 256, 256);
    cvt(W1f + (size_t)l * 131072, w1t + (size_t)l * 131072, 256, 512);
    cvt(W2f + (size_t)l * 131072, w2t + (size_t)l * 131072, 512, 256);
  }
  adj_build_kernel<<<3, 256, 0, stream>>>(adj, cnt);

  // ---- embedding ----
  embed_kernel<<<BATCH_ * SEQ_, 256, 0, stream>>>(x_enc, x_mark, W_val, W_time, b_time, emb);

  // ---- CSCM: down-proj, pyramid convs, up-proj ----
  gemm(emb, wdt, b_down, t0, BATCH_ * SEQ_, 256, 64, 0);           // [16384,64]
  {
    int L0 = 128, L1 = 32, L2 = 8;
    cscm_conv_kernel<<<(BATCH_ * L0 * 64 + 255) / 256, 256, 0, stream>>>(
        t0, (size_t)SEQ_ * 64, W_conv + 0 * 64 * 64 * 4, b_conv + 0, bn_g + 0, bn_b + 0,
        pool + 0 * 64, (size_t)168 * 64, L0);
    cscm_conv_kernel<<<(BATCH_ * L1 * 64 + 255) / 256, 256, 0, stream>>>(
        pool + 0 * 64, (size_t)168 * 64, W_conv + 1 * 64 * 64 * 4, b_conv + 64, bn_g + 64, bn_b + 64,
        pool + 128 * 64, (size_t)168 * 64, L1);
    cscm_conv_kernel<<<(BATCH_ * L2 * 64 + 255) / 256, 256, 0, stream>>>(
        pool + 128 * 64, (size_t)168 * 64, W_conv + 2 * 64 * 64 * 4, b_conv + 128, bn_g + 128, bn_b + 128,
        pool + 160 * 64, (size_t)168 * 64, L2);
  }
  gemm(pool, wut, b_up, up_out, BATCH_ * 168, 64, 256, 0);         // [5376,256]

  // ---- concat + LN -> x [21760,256] ----
  concat_ln_kernel<<<MROWS_, 256, 0, stream>>>(emb, up_out, ln_c_g, ln_c_b, x);

  // ---- 2 encoder layers ----
  for (int l = 0; l < 2; ++l) {
    const _Float16* wq = wqt + (size_t)l * 65536;
    const _Float16* wk = wkt + (size_t)l * 65536;
    const _Float16* wv = wvt + (size_t)l * 65536;
    const _Float16* wo = wot + (size_t)l * 65536;
    gemm(x, wq, bq + l * 256, qbuf, MROWS_, 256, 256, 0);
    gemm(x, wk, bk + l * 256, kbuf, MROWS_, 256, 256, 0);
    gemm(x, wv, bv + l * 256, vbuf, MROWS_, 256, 256, 0);

    int nwaves = BATCH_ * HEADS_ * NTOK_;
    attn_kernel<<<(nwaves + 7) / 8, 256, 0, stream>>>(qbuf, kbuf, vbuf, adj, cnt, attn_o);

    gemm(attn_o, wo, bo + l * 256, proj, MROWS_, 256, 256, 0);
    add_ln_kernel<<<MROWS_, 256, 0, stream>>>(x, proj, ln1_g + l * 256, ln1_b + l * 256, x);

    gemm(x, w1t + (size_t)l * 131072, b1f + l * 512, f1, MROWS_, 256, 512, 1);  // ReLU
    gemm(f1, w2t + (size_t)l * 131072, b2f + l * 256, f2, MROWS_, 512, 256, 0);
    add_ln_kernel<<<MROWS_, 256, 0, stream>>>(x, f2, ln2_g + l * 256, ln2_b + l * 256, x);
  }

  // ---- final gather + projection ----
  final_proj_kernel<<<BATCH_, 96, 0, stream>>>(x, Wp, bp, out);
}
